// MultiHeadAttentionKarpathy_38852274159862
// MI455X (gfx1250) — matrix operand order repaired and checked by reading, not compiled, not run on hardware
//
#include <hip/hip_runtime.h>

typedef __attribute__((ext_vector_type(16))) _Float16 v16h;
typedef __attribute__((ext_vector_type(8)))  _Float16 v8h;
typedef __attribute__((ext_vector_type(8)))  float    v8f;

#define BB 4
#define TT 2048
#define CC 1024
#define HH 16
#define DD 64
#define C3 (3*CC)

// ---------------- fragment loaders (CDNA5 wave32 WMMA layouts) ----------------
// A fragment (16x32 f16): lane l holds row (l&15); VGPRs 0..3 -> K = 8*(l>>4)+{0..7},
// VGPRs 4..7 -> K = 16 + 8*(l>>4)+{0..7}. Two contiguous 8-half runs per lane.
__device__ __forceinline__ v16h load_frag_a(const _Float16* p, int stride) {
  const int lane = threadIdx.x & 31;
  const int row  = lane & 15;
  const int kg   = (lane >> 4) << 3;                 // 0 or 8
  const _Float16* rp = p + row * stride;
  v8h lo = *(const v8h*)(rp + kg);
  v8h hi = *(const v8h*)(rp + 16 + kg);
  v16h a;
#pragma unroll
  for (int i = 0; i < 8; ++i) { a[i] = lo[i]; a[i + 8] = hi[i]; }
  return a;
}

// B fragment (32x16 f16) loaded from a B^T row-major source ([n][k], given stride):
// lane l holds column n=(l&15); K = 8*(l>>4)+{0..7} and 16 + 8*(l>>4)+{0..7} (matches A).
__device__ __forceinline__ v16h load_frag_bT(const _Float16* p, int stride) {
  const int lane = threadIdx.x & 31;
  const int n    = lane & 15;
  const int kg   = (lane >> 4) << 3;                 // 0 or 8
  const _Float16* cp = p + n * stride;
  v8h lo = *(const v8h*)(cp + kg);
  v8h hi = *(const v8h*)(cp + 16 + kg);
  v16h b;
#pragma unroll
  for (int i = 0; i < 8; ++i) { b[i] = lo[i]; b[i + 8] = hi[i]; }
  return b;
}

__device__ __forceinline__ v8f wmma_f16(v16h a, v16h b, v8f c) {
  return __builtin_amdgcn_wmma_f32_16x16x32_f16(false, a, false, b, (short)0, c, false, false);
}

// ---------------- f32 -> f16 conversion ----------------
__global__ void cvt_f32_to_f16_kernel(const float* __restrict__ in,
                                      _Float16* __restrict__ out, int n) {
  for (int i = blockIdx.x * blockDim.x + threadIdx.x; i < n;
       i += gridDim.x * blockDim.x)
    out[i] = (_Float16)in[i];
}

// ---------------- QKV GEMM: [8192,1024] x [1024,3072] -> Q/K/V f16 [mat][B,H,T,D] ---
#define LS 40   // LDS row stride in halves (80B: 16B-aligned, conflict-mitigating)

__global__ __launch_bounds__(256)
void qkv_gemm_kernel(const _Float16* __restrict__ xh,
                     const _Float16* __restrict__ wah,
                     const float* __restrict__ b_attn,
                     _Float16* __restrict__ qkv) {
  __shared__ _Float16 As[128 * LS];
  __shared__ _Float16 Bs[128 * LS];   // stored transposed: Bs[n][k]
  const int tid  = threadIdx.x;
  const int lane = tid & 31;
  const int wid  = tid >> 5;
  const int wm   = wid & 3;           // 4 waves along M
  const int wn   = wid >> 2;          // 2 waves along N
  const int m0   = blockIdx.y * 128;
  const int n0   = blockIdx.x * 128;

  v8f acc[2][4];
#pragma unroll
  for (int mi = 0; mi < 2; ++mi)
#pragma unroll
    for (int ni = 0; ni < 4; ++ni) acc[mi][ni] = (v8f)(0.0f);

  for (int k0 = 0; k0 < CC; k0 += 32) {
    __syncthreads();
#pragma unroll
    for (int it = 0; it < 2; ++it) {                 // A tile: 128 rows x 32 halves
      int e  = (tid + it * 256) * 8;
      int r  = e >> 5;
      int kk = e & 31;
      v8h v = *(const v8h*)(xh + (size_t)(m0 + r) * CC + k0 + kk);
      *(v8h*)(As + r * LS + kk) = v;
    }
#pragma unroll
    for (int it = 0; it < 2; ++it) {                 // B tile, transposed into LDS
      int e  = (tid + it * 256) * 8;
      int kk = e >> 7;
      int nn = e & 127;
      v8h v = *(const v8h*)(wah + (size_t)(k0 + kk) * C3 + n0 + nn);
#pragma unroll
      for (int q = 0; q < 8; ++q) Bs[(nn + q) * LS + kk] = v[q];
    }
    __syncthreads();

    v16h a[2], b[4];
#pragma unroll
    for (int mi = 0; mi < 2; ++mi)
      a[mi] = load_frag_a(As + (wm * 32 + mi * 16) * LS, LS);
#pragma unroll
    for (int ni = 0; ni < 4; ++ni)
      b[ni] = load_frag_bT(Bs + (wn * 64 + ni * 16) * LS, LS);
#pragma unroll
    for (int mi = 0; mi < 2; ++mi)
#pragma unroll
      for (int ni = 0; ni < 4; ++ni)
        acc[mi][ni] = wmma_f16(a[mi], b[ni], acc[mi][ni]);
  }

  // epilogue: bias, fold 1/sqrt(D) into Q, scatter into [mat][(b*H+h)*T + t]*D + d
  const size_t plane = (size_t)TT * DD;
#pragma unroll
  for (int mi = 0; mi < 2; ++mi)
#pragma unroll
    for (int ni = 0; ni < 4; ++ni) {
      int col = n0 + wn * 64 + ni * 16 + (lane & 15);
      int mat = col >> 10;
      int cc  = col & 1023;
      int h   = cc >> 6, d = cc & 63;
      float bias = b_attn[col];
#pragma unroll
      for (int i = 0; i < 8; ++i) {
        int row = m0 + wm * 32 + mi * 16 + i + ((lane >> 4) << 3);
        int bb  = row >> 11, t = row & 2047;
        float v = acc[mi][ni][i] + bias;
        if (mat == 0) v *= 0.125f;                   // 1/sqrt(64)
        qkv[((size_t)mat * (BB * HH) + (size_t)(bb * HH + h)) * plane +
            (size_t)t * DD + d] = (_Float16)v;
      }
    }
}

// ---------------- Flash attention: per-wave 16-query tile, 32-key blocks ----------
__global__ __launch_bounds__(128)
void attn_kernel(const _Float16* __restrict__ qkv, _Float16* __restrict__ yh) {
  __shared__ _Float16 lds[4 * (64 * LS + 16 * LS)];  // per wave: V^T (64xLS) + P (16xLS)
  const int tid  = threadIdx.x;
  const int lane = tid & 31;
  const int w    = tid >> 5;
  _Float16* Vt = lds + w * (64 * LS + 16 * LS);
  _Float16* Pt = Vt + 64 * LS;

  const int bh = blockIdx.y;
  const int qb = blockIdx.x * 64 + w * 16;           // query tile base
  const size_t plane = (size_t)TT * DD;
  const _Float16* qp = qkv + (size_t)bh * plane;
  const _Float16* kp = qkv + (size_t)(BB * HH) * plane + (size_t)bh * plane;
  const _Float16* vp = qkv + (size_t)(2 * BB * HH) * plane + (size_t)bh * plane;

  const v16h aq0 = load_frag_a(qp + (size_t)qb * DD, DD);       // d 0..31
  const v16h aq1 = load_frag_a(qp + (size_t)qb * DD + 32, DD);  // d 32..63

  float mrow[8], lrow[8];
  v8f o[4];
#pragma unroll
  for (int i = 0; i < 8; ++i) { mrow[i] = -1e30f; lrow[i] = 0.0f; }
#pragma unroll
  for (int t = 0; t < 4; ++t) o[t] = (v8f)(0.0f);

  const int nkb = qb / 32 + 1;                       // causal block count
  for (int kb = 0; kb < nkb; ++kb) {
    const int kbase = kb * 32;

    // stage V^T in LDS: Vt[d][key], paired-key 32-bit stores
    {
      const int a2 = lane & 15;                      // key pair
      const int dh = (lane >> 4) * 32;               // d half
      const _Float16* r0 = vp + (size_t)(kbase + 2 * a2) * DD + dh;
      const _Float16* r1 = r0 + DD;
      v8h x0[4], x1[4];
#pragma unroll
      for (int q = 0; q < 4; ++q) {
        x0[q] = *(const v8h*)(r0 + q * 8);
        x1[q] = *(const v8h*)(r1 + q * 8);
      }
#pragma unroll
      for (int d = 0; d < 32; ++d) {
        union { _Float16 h[2]; unsigned u; } pr;
        pr.h[0] = x0[d >> 3][d & 7];
        pr.h[1] = x1[d >> 3][d & 7];
        *(unsigned*)(Vt + (dh + d) * LS + 2 * a2) = pr.u;
      }
    }

    // S = Q K^T  (two 16-key subtiles; K global layout is already B^T row-major)
    v8f s0 = (v8f)(0.0f), s1 = (v8f)(0.0f);
    s0 = wmma_f16(aq0, load_frag_bT(kp + (size_t)kbase * DD, DD), s0);
    s0 = wmma_f16(aq1, load_frag_bT(kp + (size_t)kbase * DD + 32, DD), s0);
    s1 = wmma_f16(aq0, load_frag_bT(kp + (size_t)(kbase + 16) * DD, DD), s1);
    s1 = wmma_f16(aq1, load_frag_bT(kp + (size_t)(kbase + 16) * DD + 32, DD), s1);

    if (kb == nkb - 1) {                             // causal mask on diagonal block
      const int c0 = kbase + (lane & 15);
#pragma unroll
      for (int i = 0; i < 8; ++i) {
        int row = qb + i + ((lane >> 4) << 3);
        if (c0 > row)      s0[i] = -1e30f;
        if (c0 + 16 > row) s1[i] = -1e30f;
      }
    }

    // online softmax (row groups live in 16-lane halves; xor<16 stays inside)
    float fac[8];
#pragma unroll
    for (int i = 0; i < 8; ++i) {
      float mx = fmaxf(s0[i], s1[i]);
#pragma unroll
      for (int off = 8; off >= 1; off >>= 1)
        mx = fmaxf(mx, __shfl_xor(mx, off, 32));
      float mn = fmaxf(mrow[i], mx);
      float f  = __expf(mrow[i] - mn);
      float p0 = __expf(s0[i] - mn);
      float p1 = __expf(s1[i] - mn);
      float rs = p0 + p1;
#pragma unroll
      for (int off = 8; off >= 1; off >>= 1)
        rs += __shfl_xor(rs, off, 32);
      lrow[i] = lrow[i] * f + rs;
      mrow[i] = mn;
      fac[i]  = f;
      const int prow = i + ((lane >> 4) << 3);       // C-layout -> LDS (row-major P)
      Pt[prow * LS + (lane & 15)]      = (_Float16)p0;
      Pt[prow * LS + 16 + (lane & 15)] = (_Float16)p1;
    }
#pragma unroll
    for (int t = 0; t < 4; ++t)
#pragma unroll
      for (int i = 0; i < 8; ++i) o[t][i] *= fac[i];

    // P (A-layout via LDS) x V^T tiles
    const v16h pa = load_frag_a(Pt, LS);
#pragma unroll
    for (int t = 0; t < 4; ++t)
      o[t] = wmma_f16(pa, load_frag_bT(Vt + t * 16 * LS, LS), o[t]);
  }

  // finalize and store y as f16 [B,T,C]
  float inv[8];
#pragma unroll
  for (int i = 0; i < 8; ++i) inv[i] = 1.0f / lrow[i];
  const int b = bh >> 4, h = bh & 15;
#pragma unroll
  for (int t = 0; t < 4; ++t)
#pragma unroll
    for (int i = 0; i < 8; ++i) {
      int row = qb + i + ((lane >> 4) << 3);
      int col = h * 64 + t * 16 + (lane & 15);
      yh[((size_t)b * TT + row) * CC + col] = (_Float16)(o[t][i] * inv[i]);
    }
}

// ---------------- Projection GEMM: [8192,1024] x [1024,1024] + bias -> f32 out -----
__global__ __launch_bounds__(256)
void proj_gemm_kernel(const _Float16* __restrict__ yh,
                      const _Float16* __restrict__ wph,
                      const float* __restrict__ b_proj,
                      float* __restrict__ out) {
  __shared__ _Float16 As[128 * LS];
  __shared__ _Float16 Bs[128 * LS];
  const int tid  = threadIdx.x;
  const int lane = tid & 31;
  const int wid  = tid >> 5;
  const int wm   = wid & 3;
  const int wn   = wid >> 2;
  const int m0   = blockIdx.y * 128;
  const int n0   = blockIdx.x * 128;

  v8f acc[2][4];
#pragma unroll
  for (int mi = 0; mi < 2; ++mi)
#pragma unroll
    for (int ni = 0; ni < 4; ++ni) acc[mi][ni] = (v8f)(0.0f);

  for (int k0 = 0; k0 < CC; k0 += 32) {
    __syncthreads();
#pragma unroll
    for (int it = 0; it < 2; ++it) {
      int e  = (tid + it * 256) * 8;
      int r  = e >> 5;
      int kk = e & 31;
      v8h v = *(const v8h*)(yh + (size_t)(m0 + r) * CC + k0 + kk);
      *(v8h*)(As + r * LS + kk) = v;
    }
#pragma unroll
    for (int it = 0; it < 2; ++it) {
      int e  = (tid + it * 256) * 8;
      int kk = e >> 7;
      int nn = e & 127;
      v8h v = *(const v8h*)(wph + (size_t)(k0 + kk) * CC + n0 + nn);
#pragma unroll
      for (int q = 0; q < 8; ++q) Bs[(nn + q) * LS + kk] = v[q];
    }
    __syncthreads();

    v16h a[2], b[4];
#pragma unroll
    for (int mi = 0; mi < 2; ++mi)
      a[mi] = load_frag_a(As + (wm * 32 + mi * 16) * LS, LS);
#pragma unroll
    for (int ni = 0; ni < 4; ++ni)
      b[ni] = load_frag_bT(Bs + (wn * 64 + ni * 16) * LS, LS);
#pragma unroll
    for (int mi = 0; mi < 2; ++mi)
#pragma unroll
      for (int ni = 0; ni < 4; ++ni)
        acc[mi][ni] = wmma_f16(a[mi], b[ni], acc[mi][ni]);
  }

#pragma unroll
  for (int mi = 0; mi < 2; ++mi)
#pragma unroll
    for (int ni = 0; ni < 4; ++ni) {
      int col = n0 + wn * 64 + ni * 16 + (lane & 15);
      float bias = b_proj[col];
#pragma unroll
      for (int i = 0; i < 8; ++i) {
        int row = m0 + wm * 32 + mi * 16 + i + ((lane >> 4) << 3);
        out[(size_t)row * CC + col] = acc[mi][ni][i] + bias;
      }
    }
}

// ---------------- host launch ----------------
extern "C" void kernel_launch(void* const* d_in, const int* in_sizes, int n_in,
                              void* d_out, int out_size, void* d_ws, size_t ws_size,
                              hipStream_t stream) {
  const float* x      = (const float*)d_in[0];   // [4,2048,1024]
  const float* W_attn = (const float*)d_in[1];   // [1024,3072]
  const float* b_attn = (const float*)d_in[2];   // [3072]
  const float* W_proj = (const float*)d_in[3];   // [1024,1024]
  const float* b_proj = (const float*)d_in[4];   // [1024]
  float* out = (float*)d_out;                    // [4,2048,1024] f32

  const size_t nX  = (size_t)BB * TT * CC;       // 8388608
  const size_t nWA = (size_t)CC * C3;            // 3145728
  const size_t nWP = (size_t)CC * CC;            // 1048576
  const size_t nQKV = 3 * nX;

  char* ws = (char*)d_ws;
  _Float16* xh   = (_Float16*)(ws);
  _Float16* wah  = (_Float16*)(ws + 2 * nX);
  _Float16* wph  = (_Float16*)(ws + 2 * (nX + nWA));
  _Float16* qkvh = (_Float16*)(ws + 2 * (nX + nWA + nWP));
  _Float16* yhb  = (_Float16*)(ws + 2 * (nX + nWA + nWP + nQKV));

  cvt_f32_to_f16_kernel<<<2048, 256, 0, stream>>>(x, xh, (int)nX);
  cvt_f32_to_f16_kernel<<<2048, 256, 0, stream>>>(W_attn, wah, (int)nWA);
  cvt_f32_to_f16_kernel<<<1024, 256, 0, stream>>>(W_proj, wph, (int)nWP);

  qkv_gemm_kernel<<<dim3(C3 / 128, (BB * TT) / 128), 256, 0, stream>>>(
      xh, wah, b_attn, qkvh);

  attn_kernel<<<dim3(TT / 64, BB * HH), 128, 0, stream>>>(qkvh, yhb);

  proj_gemm_kernel<<<dim3(CC / 128, (BB * TT) / 128), 256, 0, stream>>>(
      yhb, wph, b_proj, out);
}
